// MoELayer_43654047597179
// MI455X (gfx1250) — compile-verified, matrix operand order checked by reading, not run on hardware
//
#include <hip/hip_runtime.h>

// MoE FFN (uniform split): per expert  h = silu(x@w1) * (x@w3);  out = h@w2
// E=8, D=1024, H=4096, 4096 tokens/expert. Compute-bound (~824 GFLOP vs ~29us
// of HBM traffic at 23.3 TB/s) -> everything on v_wmma_f32_16x16x32_bf16 with
// f32->bf16 downconvert during LDS staging. Double-buffered LDS software
// pipeline; kernel 2's bf16 h tile is DMA'd by the Tensor Data Mover
// (tensor_load_to_lds) with LDS padding matching our 96B bank-swizzled stride.

#define NE 8
#define DD 1024
#define HH 4096
#define ME 4096          // tokens per expert
#define BM 128
#define BN 128
#define BK 32
#define LDSTR 48         // ushort stride (32 data + 16 pad) = 96B, 16B aligned

typedef __attribute__((ext_vector_type(16))) __bf16        v16bf;
typedef __attribute__((ext_vector_type(8)))  float         v8f;
typedef __attribute__((ext_vector_type(4)))  float         fx4;
typedef __attribute__((ext_vector_type(4)))  unsigned int  u32x4;
typedef __attribute__((ext_vector_type(8)))  int           i32x8;
typedef __attribute__((ext_vector_type(4)))  int           i32x4;

struct U128x2 { uint4 lo, hi; };

__device__ __forceinline__ unsigned short f2bf(float f) {
  return __builtin_bit_cast(unsigned short, (__bf16)f);
}

// Fragment = 16 bf16 per lane: two contiguous 16B chunks at kb and kb+16
// (16-bit A/B operand VGPR layout, wave32).
__device__ __forceinline__ v16bf load_frag(const unsigned short* base) {
  U128x2 t;
  t.lo = *reinterpret_cast<const uint4*>(base);
  t.hi = *reinterpret_cast<const uint4*>(base + 16);
  return __builtin_bit_cast(v16bf, t);
}

__device__ __forceinline__ unsigned long long pack4bf(fx4 f) {
  return (unsigned long long)f2bf(f.x)        |
         ((unsigned long long)f2bf(f.y) << 16) |
         ((unsigned long long)f2bf(f.z) << 32) |
         ((unsigned long long)f2bf(f.w) << 48);
}

// ---------------- Kernel 1: gate/up fused, writes bf16 h to workspace --------
__global__ __launch_bounds__(256) void moe_gateup(
    const float* __restrict__ x, const float* __restrict__ w1,
    const float* __restrict__ w3, unsigned short* __restrict__ hout) {
  __shared__ unsigned short xs [2][BM * LDSTR];
  __shared__ unsigned short w1s[2][BN * LDSTR];
  __shared__ unsigned short w3s[2][BN * LDSTR];

  const int n0   = blockIdx.x * BN;
  const int m0   = blockIdx.y * BM;
  const int e    = blockIdx.z;
  const int tid  = threadIdx.x;
  const int lane = tid & 31;
  const int wave = tid >> 5;
  const int waveM = wave & 3;          // 4 strips of 32 rows
  const int waveN = wave >> 2;         // 2 strips of 64 cols
  const int l15   = lane & 15;
  const int kb    = (lane >> 4) << 3;  // 0 or 8
  const int rowadd = (lane >> 4) << 3; // C/D layout: +8 rows for upper lanes

  const float* xblk  = x  + (size_t)(e * ME + m0) * DD;
  const float* w1blk = w1 + (size_t)e * DD * HH + n0;
  const float* w3blk = w3 + (size_t)e * DD * HH + n0;

  v8f accg[2][4], accu[2][4];
  const v8f vzero = {0.f,0.f,0.f,0.f,0.f,0.f,0.f,0.f};
#pragma unroll
  for (int i = 0; i < 2; ++i)
#pragma unroll
    for (int j = 0; j < 4; ++j) { accg[i][j] = vzero; accu[i][j] = vzero; }

  fx4 fx[4], f1[4], f3[4];   // in-flight staging registers (next k-tile)

  auto load_tiles = [&](int k0) {
#pragma unroll
    for (int i = 0; i < 4; ++i) {
      int v = tid + i * 256;                       // 1024 float4s (x tile)
      fx[i] = __builtin_nontemporal_load(
          reinterpret_cast<const fx4*>(xblk + (size_t)(v >> 3) * DD + k0) + (v & 7));
    }
#pragma unroll
    for (int i = 0; i < 4; ++i) {
      int v = tid + i * 256;                       // 1024 float4s (w tiles)
      const fx4* p1 = reinterpret_cast<const fx4*>(w1blk + (size_t)(k0 + (v >> 5)) * HH);
      const fx4* p3 = reinterpret_cast<const fx4*>(w3blk + (size_t)(k0 + (v >> 5)) * HH);
      f1[i] = __builtin_nontemporal_load(p1 + (v & 31));
      f3[i] = __builtin_nontemporal_load(p3 + (v & 31));
    }
  };

  auto store_tiles = [&](int b) {
#pragma unroll
    for (int i = 0; i < 4; ++i) {                  // x tile row-major [m][k]
      int v = tid + i * 256;
      int r = v >> 3, c4 = v & 7;
      *reinterpret_cast<unsigned long long*>(&xs[b][r * LDSTR + c4 * 4]) = pack4bf(fx[i]);
    }
#pragma unroll
    for (int i = 0; i < 4; ++i) {                  // w tiles transposed [n][k]
      int v = tid + i * 256;
      int kr = v >> 5, nb = (v & 31) * 4;
      w1s[b][(nb + 0) * LDSTR + kr] = f2bf(f1[i].x);
      w1s[b][(nb + 1) * LDSTR + kr] = f2bf(f1[i].y);
      w1s[b][(nb + 2) * LDSTR + kr] = f2bf(f1[i].z);
      w1s[b][(nb + 3) * LDSTR + kr] = f2bf(f1[i].w);
      w3s[b][(nb + 0) * LDSTR + kr] = f2bf(f3[i].x);
      w3s[b][(nb + 1) * LDSTR + kr] = f2bf(f3[i].y);
      w3s[b][(nb + 2) * LDSTR + kr] = f2bf(f3[i].z);
      w3s[b][(nb + 3) * LDSTR + kr] = f2bf(f3[i].w);
    }
  };

  auto compute = [&](int b) {
    v16bf a[2];
#pragma unroll
    for (int i = 0; i < 2; ++i)
      a[i] = load_frag(&xs[b][(waveM * 32 + i * 16 + l15) * LDSTR + kb]);
#pragma unroll
    for (int j = 0; j < 4; ++j) {
      v16bf b1 = load_frag(&w1s[b][(waveN * 64 + j * 16 + l15) * LDSTR + kb]);
      v16bf b3 = load_frag(&w3s[b][(waveN * 64 + j * 16 + l15) * LDSTR + kb]);
#pragma unroll
      for (int i = 0; i < 2; ++i) {
        accg[i][j] = __builtin_amdgcn_wmma_f32_16x16x32_bf16(
            false, a[i], false, b1, (short)0, accg[i][j], false, false);
        accu[i][j] = __builtin_amdgcn_wmma_f32_16x16x32_bf16(
            false, a[i], false, b3, (short)0, accu[i][j], false, false);
      }
    }
  };

  // software pipeline: prologue stages tile 0, loop overlaps global loads of
  // tile i+1 with WMMA on tile i (double-buffered LDS, 1 barrier/iter)
  load_tiles(0);
  store_tiles(0);
  __syncthreads();
  const int NK = DD / BK;                          // 32
  for (int ki = 0; ki < NK; ++ki) {
    const int nxt = ki + 1;
    if (nxt < NK) load_tiles(nxt * BK);
    compute(ki & 1);
    if (nxt < NK) store_tiles(nxt & 1);
    __syncthreads();
  }

  // epilogue: silu(gate)*up -> bf16 h
#pragma unroll
  for (int i = 0; i < 2; ++i) {
#pragma unroll
    for (int j = 0; j < 4; ++j) {
#pragma unroll
      for (int r = 0; r < 8; ++r) {
        float g = accg[i][j][r];
        float u = accu[i][j][r];
        float s = (g / (1.0f + __expf(-g))) * u;
        int mrow = m0 + waveM * 32 + i * 16 + r + rowadd;
        int ncol = n0 + waveN * 64 + j * 16 + l15;
        hout[((size_t)e * ME + mrow) * HH + ncol] = f2bf(s);
      }
    }
  }
}

// ---------------- Kernel 2: down projection out = h @ w2 (f32 out) ----------
// h tile (pure bf16 copy) is moved by the Tensor Data Mover: 128x32 2-byte 2D
// tile, TDM pad_enable inserts 8 DWORDs (32B) after every 16 DWORDs (64B row)
// -> 96B LDS row stride == LDSTR shorts, matching the compute-side layout.
__global__ __launch_bounds__(256) void moe_down(
    const unsigned short* __restrict__ h, const float* __restrict__ w2,
    float* __restrict__ out) {
  __shared__ unsigned short hs [2][BM * LDSTR];
  __shared__ unsigned short w2s[2][BN * LDSTR];

  const int n0   = blockIdx.x * BN;    // over D
  const int m0   = blockIdx.y * BM;
  const int e    = blockIdx.z;
  const int tid  = threadIdx.x;
  const int lane = tid & 31;
  const int wave = tid >> 5;
  const int waveM = wave & 3;
  const int waveN = wave >> 2;
  const int l15   = lane & 15;
  const int kb    = (lane >> 4) << 3;
  const int rowadd = (lane >> 4) << 3;

  const unsigned short* hblk = h + (size_t)(e * ME + m0) * HH;
  const float* w2blk = w2 + (size_t)e * HH * DD + n0;

  v8f acc[2][4];
  const v8f vzero = {0.f,0.f,0.f,0.f,0.f,0.f,0.f,0.f};
#pragma unroll
  for (int i = 0; i < 2; ++i)
#pragma unroll
    for (int j = 0; j < 4; ++j) acc[i][j] = vzero;

  fx4 fw[4];

  // One wave per block drives the TDM (EXEC-independent, SGPR descriptors,
  // async under TENSORcnt).  amdgpu-toolchain lane: 6-arg builtin form.
  auto issue_tdm = [&](int k0, int b) {
    if (wave == 0) {
      unsigned long long ga =
          (unsigned long long)(uintptr_t)hblk + (unsigned long long)k0 * 2ull;
      unsigned int ldsoff = (unsigned int)(uintptr_t)&hs[b][0];
      // D# group0: count=1 | lds_addr | global_addr[56:0] | type=2
      u32x4 g0 = { 1u, ldsoff, (unsigned int)ga,
                   (unsigned int)(((ga >> 32) & 0x01FFFFFFull) | 0x80000000u) };
      // D# group1: data_size=2B, pad_enable, pad_interval=16 DW, pad_amount=8 DW,
      //            tensor_dim0=HH, tensor_dim1=ME, tile_dim0=32, tile_dim1=128,
      //            tensor_dim0_stride=HH
      i32x8 g1 = { (int)((1u << 16) | (1u << 20) | (3u << 22) | (7u << 25)),
                   (int)((HH & 0xFFFFu) << 16),
                   (int)(((HH >> 16) & 0xFFFFu) | ((ME & 0xFFFFu) << 16)),
                   (int)(((ME >> 16) & 0xFFFFu) | (32u << 16)),
                   (int)(128u),
                   (int)(HH),
                   0, 0 };
      i32x4 g2 = { 0, 0, 0, 0 };
      i32x4 g3 = { 0, 0, 0, 0 };
      i32x8 g4 = { 0, 0, 0, 0, 0, 0, 0, 0 };
      __builtin_amdgcn_tensor_load_to_lds(g0, g1, g2, g3, g4, 0);
    }
  };

  auto load_w = [&](int k0) {
#pragma unroll
    for (int i = 0; i < 4; ++i) {
      int v = tid + i * 256;
      const fx4* p = reinterpret_cast<const fx4*>(w2blk + (size_t)(k0 + (v >> 5)) * DD);
      fw[i] = __builtin_nontemporal_load(p + (v & 31));
    }
  };

  auto store_w = [&](int b) {
#pragma unroll
    for (int i = 0; i < 4; ++i) {                  // transposed [n][k]
      int v = tid + i * 256;
      int kr = v >> 5, nb = (v & 31) * 4;
      w2s[b][(nb + 0) * LDSTR + kr] = f2bf(fw[i].x);
      w2s[b][(nb + 1) * LDSTR + kr] = f2bf(fw[i].y);
      w2s[b][(nb + 2) * LDSTR + kr] = f2bf(fw[i].z);
      w2s[b][(nb + 3) * LDSTR + kr] = f2bf(fw[i].w);
    }
  };

  auto compute = [&](int b) {
    v16bf a[2];
#pragma unroll
    for (int i = 0; i < 2; ++i)
      a[i] = load_frag(&hs[b][(waveM * 32 + i * 16 + l15) * LDSTR + kb]);
#pragma unroll
    for (int j = 0; j < 4; ++j) {
      v16bf bfr = load_frag(&w2s[b][(waveN * 64 + j * 16 + l15) * LDSTR + kb]);
#pragma unroll
      for (int i = 0; i < 2; ++i) {
        acc[i][j] = __builtin_amdgcn_wmma_f32_16x16x32_bf16(
            false, a[i], false, bfr, (short)0, acc[i][j], false, false);
      }
    }
  };

  // pipeline: TDM + w2 global loads for tile i+1 overlap WMMA on tile i
  issue_tdm(0, 0);
  load_w(0);
  store_w(0);
  __builtin_amdgcn_s_wait_tensorcnt(0);
  __syncthreads();

  const int NK = HH / BK;                          // 128
  for (int ki = 0; ki < NK; ++ki) {
    const int nxt = ki + 1;
    if (nxt < NK) {
      issue_tdm(nxt * BK, nxt & 1);
      load_w(nxt * BK);
    }
    compute(ki & 1);
    if (nxt < NK) store_w(nxt & 1);
    __builtin_amdgcn_s_wait_tensorcnt(0);
    __syncthreads();
  }

#pragma unroll
  for (int i = 0; i < 2; ++i) {
#pragma unroll
    for (int j = 0; j < 4; ++j) {
#pragma unroll
      for (int r = 0; r < 8; ++r) {
        int mrow = m0 + waveM * 32 + i * 16 + r + rowadd;
        int ncol = n0 + waveN * 64 + j * 16 + l15;
        out[((size_t)e * ME + mrow) * DD + ncol] = acc[i][j][r];
      }
    }
  }
}

extern "C" void kernel_launch(void* const* d_in, const int* in_sizes, int n_in,
                              void* d_out, int out_size, void* d_ws, size_t ws_size,
                              hipStream_t stream) {
  const float* x  = (const float*)d_in[0];
  const float* w1 = (const float*)d_in[1];
  const float* w2 = (const float*)d_in[2];
  const float* w3 = (const float*)d_in[3];
  // d_in[4] = splits (uniform T/E, compile-time here)

  unsigned short* hbuf = (unsigned short*)d_ws;   // [T, H] bf16 = 256 MiB

  dim3 g1(HH / BN, ME / BM, NE);
  moe_gateup<<<g1, 256, 0, stream>>>(x, w1, w3, hbuf);

  dim3 g2(DD / BN, ME / BM, NE);
  moe_down<<<g2, 256, 0, stream>>>(hbuf, w2, (float*)d_out);
}
